// SimpleLSTM_28922309771594
// MI455X (gfx1250) — matrix operand order repaired and checked by reading, not compiled, not run on hardware
//
#include <hip/hip_runtime.h>
#include <hip/hip_bf16.h>

// LSTM: B=64, T=1024, I=512, H=512. Output h_n [64,512] fp32.
#define B_ 64
#define T_ 1024
#define I_ 512
#define H_ 512
#define G_ 2048   // 4*H
#define NWG 32    // workgroups in recurrent scan (each owns 16 h-columns)

typedef __bf16 bf16;
typedef __attribute__((ext_vector_type(16))) __bf16 v16bf;
typedef __attribute__((ext_vector_type(8)))  __bf16 v8bf;
typedef __attribute__((ext_vector_type(8)))  float  v8f;

union Frag { v16bf v; v8bf h[2]; };

static __device__ __forceinline__ v8f wmma_bf16(v16bf a, v16bf b, v8f c) {
  // D = A(16x32 bf16) * B(32x16 bf16) + C(16x16 f32)
  return __builtin_amdgcn_wmma_f32_16x16x32_bf16(
      /*neg_a=*/false, a, /*neg_b=*/false, b,
      /*c_mod=*/(short)0, c, /*reuse_a=*/false, /*reuse_b=*/false);
}

static __device__ __forceinline__ float sigmoidf_fast(float x) {
  return 1.0f / (1.0f + __expf(-x));
}
static __device__ __forceinline__ float tanhf_fast(float x) {
  // tanh(x) = 2*sigmoid(2x) - 1  (v_exp_f32 + v_rcp_f32, no ocml call)
  return __builtin_fmaf(2.0f, sigmoidf_fast(2.0f * x), -1.0f);
}

// A fragment (16x32 bf16): lane = row M (0..15), half selects K {0-15} / {16-31}
#define LOAD_A(dst, rowbase, kt, half)                        \
  do {                                                        \
    const bf16* _ap = (rowbase) + (kt) * 32 + 16 * (half);    \
    (dst).h[0] = *(const v8bf*)(_ap);                         \
    (dst).h[1] = *(const v8bf*)(_ap + 8);                     \
  } while (0)

// B fragment from fragment-major LDS: 32B contiguous per lane
#define LOAD_B(dst, ldsB, kt, nt, lane)                                          \
  do {                                                                           \
    const bf16* _bp = (ldsB) + ((size_t)(((kt) * 4 + (nt)) * 32 + (lane))) * 16; \
    (dst).h[0] = *(const v8bf*)(_bp);                                            \
    (dst).h[1] = *(const v8bf*)(_bp + 8);                                        \
  } while (0)

// ---------------------------------------------------------------- kernel 0
__global__ void lstm_cvt_x(const float* __restrict__ x, bf16* __restrict__ xb, int n) {
  for (int i = blockIdx.x * blockDim.x + threadIdx.x; i < n; i += gridDim.x * blockDim.x)
    xb[i] = (bf16)x[i];
}

__global__ void lstm_init_bar(int* __restrict__ bar) {
  if (threadIdx.x < 8) bar[threadIdx.x] = 0;
}

// ---------------------------------------------------------------- kernel 1
// xg[m, g] = sum_k xb[m,k] * Wih[g,k] + (bih[g]+bhh[g]);  m = b*T + t. M=65536, N=2048, K=512.
// Block: 256 threads = 8 waves; tile 256(M) x 64(N); wave tile 32x64 (2 A-frags x 4 B-frags,
// 8 WMMAs per k-step). kt loop fully unrolled: all fragment locals stay in VGPRs and the
// scheduler hoists load clauses across iterations (no dynamically-indexed arrays!).
__global__ void lstm_xg_gemm(const bf16* __restrict__ xb, const float* __restrict__ Wih,
                             const float* __restrict__ bih, const float* __restrict__ bhh,
                             float* __restrict__ xg) {
  extern __shared__ char smem[];
  bf16* ldsB = (bf16*)smem;                    // 16 kt * 4 nt * 32 lanes * 16 elems = 64 KiB
  const int mbase = blockIdx.x * 256;
  const int nbase = blockIdx.y * 64;
  const int tid = threadIdx.x;

  // Fill B fragments (fragment-major): lane slot s -> 16 contiguous K values of one column.
  // B[k][n] = Wih[n][k]; Wih rows are contiguous in k -> contiguous 64B fp32 read per slot.
  for (int s = tid; s < 16 * 4 * 32; s += 256) {
    const int lane_s = s & 31;
    const int nt     = (s >> 5) & 3;
    const int kt     = s >> 7;
    const int g = nbase + nt * 16 + (lane_s & 15);
    const int k = kt * 32 + 16 * (lane_s >> 4);
    const float* wp = Wih + (size_t)g * I_ + k;
    bf16* dst = ldsB + (size_t)s * 16;
#pragma unroll
    for (int j = 0; j < 16; ++j) dst[j] = (bf16)wp[j];
  }
  __syncthreads();

  const int wave = tid >> 5, lane = tid & 31;
  const int half = lane >> 4, row = lane & 15;
  const bf16* arow0 = xb + (size_t)(mbase + wave * 32 + row) * I_;       // m-frag 0
  const bf16* arow1 = arow0 + (size_t)16 * I_;                           // m-frag 1

  v8f acc[2][4];   // only constant indices after full unroll -> VGPRs
#pragma unroll
  for (int q = 0; q < 2; ++q)
#pragma unroll
    for (int p = 0; p < 4; ++p)
#pragma unroll
      for (int r = 0; r < 8; ++r) acc[q][p][r] = 0.0f;

#pragma unroll
  for (int kt = 0; kt < 16; ++kt) {
    Frag a0, a1;
    LOAD_A(a0, arow0, kt, half);
    LOAD_A(a1, arow1, kt, half);
#pragma unroll
    for (int nt = 0; nt < 4; ++nt) {
      Frag b;
      LOAD_B(b, ldsB, kt, nt, lane);
      acc[0][nt] = wmma_bf16(a0.v, b.v, acc[0][nt]);
      acc[1][nt] = wmma_bf16(a1.v, b.v, acc[1][nt]);
    }
  }

  // C/D layout: VGPR r -> M = r + 8*half, N = lane&15
#pragma unroll
  for (int nt = 0; nt < 4; ++nt) {
    const int g = nbase + nt * 16 + row;
    const float bias = bih[g] + bhh[g];
#pragma unroll
    for (int q = 0; q < 2; ++q) {
#pragma unroll
      for (int r = 0; r < 8; ++r) {
        const int mo = mbase + wave * 32 + q * 16 + r + 8 * half;
        xg[(size_t)mo * G_ + g] = acc[q][nt][r] + bias;
      }
    }
  }
}

// ---------------------------------------------------------------- grid barrier
static __device__ __forceinline__ void grid_barrier(int* __restrict__ bar, int tid) {
  __syncthreads();
  if (tid == 0) {
    const int gen = atomicAdd(&bar[1], 0);
    __threadfence();
    const int arrived = atomicAdd(&bar[0], 1);
    if (arrived == NWG - 1) {
      atomicExch(&bar[0], 0);
      __threadfence();
      atomicAdd(&bar[1], 1);
    } else {
      while (atomicAdd(&bar[1], 0) == gen) { __builtin_amdgcn_s_sleep(1); }
    }
  }
  __syncthreads();
  __threadfence();
}

// ---------------------------------------------------------------- kernel 2
// Persistent recurrent scan. 32 WGs x 256 threads. WG wg owns h-columns [wg*16, wg*16+16).
// Per step: gates(64 x 64) = h(64x512,bf16) @ Whh-cols + xg_t; then elementwise LSTM.
__global__ void lstm_scan(const float* __restrict__ Whh, const float* __restrict__ xg,
                          bf16* __restrict__ hbuf, float* __restrict__ out,
                          int* __restrict__ bar) {
  extern __shared__ char smem[];
  bf16*  ldsB   = (bf16*)smem;                  // 64 KiB: W_hh fragments, loaded once
  float* gatesL = (float*)(smem + 65536);       // 64 x 68 f32 (padded)
  const int wg  = blockIdx.x;
  const int j0  = wg * 16;
  const int tid = threadIdx.x;

  // Preload B fragments from W_hh (nt = gate block i/f/g/o, 16 columns j0..j0+15).
  for (int s = tid; s < 16 * 4 * 32; s += 256) {
    const int lane_s = s & 31;
    const int nt     = (s >> 5) & 3;
    const int kt     = s >> 7;
    const int g = nt * H_ + j0 + (lane_s & 15);
    const int k = kt * 32 + 16 * (lane_s >> 4);
    const float* wp = Whh + (size_t)g * H_ + k;
    bf16* dst = ldsB + (size_t)s * 16;
#pragma unroll
    for (int j = 0; j < 16; ++j) dst[j] = (bf16)wp[j];
  }
  // Zero both h double-buffers on our column slice (h0 = 0).
  for (int e = tid; e < B_ * 16; e += 256) {
    const int b = e >> 4, j = j0 + (e & 15);
    hbuf[(size_t)b * H_ + j]             = (bf16)0.0f;
    hbuf[(size_t)(B_ * H_) + b * H_ + j] = (bf16)0.0f;
  }
  __threadfence();
  grid_barrier(bar, tid);

  const int wave = tid >> 5, lane = tid & 31;
  const int half = lane >> 4, row = lane & 15;
  const int mtile = wave & 3;        // 4 row-tiles of 16 batch rows
  const int npair = wave >> 2;       // 2 waves share row-tile, split 64 cols into 2x32

  float c[4];
#pragma unroll
  for (int q = 0; q < 4; ++q) c[q] = 0.0f;

  for (int t = 0; t < T_; ++t) {
    const bf16* hcur  = hbuf + (size_t)(t & 1) * (B_ * H_);
    bf16*       hnext = hbuf + (size_t)((t + 1) & 1) * (B_ * H_);

    // Issue xg gate loads for this step up-front so they overlap the GEMM.
    float xv[4][4];
#pragma unroll
    for (int q = 0; q < 4; ++q) {
      const int e  = tid + q * 256;
      const int b  = e >> 4;
      const int jl = e & 15;
      const size_t xoff = ((size_t)(b * T_ + t)) * G_ + j0 + jl;
#pragma unroll
      for (int gb = 0; gb < 4; ++gb) xv[q][gb] = xg[xoff + (size_t)gb * H_];
    }

    v8f acc[2];
#pragma unroll
    for (int p = 0; p < 2; ++p)
#pragma unroll
      for (int r = 0; r < 8; ++r) acc[p][r] = 0.0f;

    const bf16* arow = hcur + (size_t)(mtile * 16 + row) * H_;  // A rows = batch
#pragma unroll
    for (int kt = 0; kt < 16; ++kt) {
      Frag a;
      LOAD_A(a, arow, kt, half);
#pragma unroll
      for (int p = 0; p < 2; ++p) {
        Frag b;
        LOAD_B(b, ldsB, kt, npair * 2 + p, lane);
        acc[p] = wmma_bf16(a.v, b.v, acc[p]);
      }
    }

    // Stage gate tiles to LDS so each thread can gather its i/f/g/o quadruple.
#pragma unroll
    for (int p = 0; p < 2; ++p) {
      const int nt = npair * 2 + p;
#pragma unroll
      for (int r = 0; r < 8; ++r) {
        const int b_ix = mtile * 16 + r + 8 * half;
        gatesL[b_ix * 68 + nt * 16 + row] = acc[p][r];
      }
    }
    __syncthreads();

    // Elementwise LSTM update: 1024 (b, j) cells / 256 threads = 4 each (fixed ownership).
#pragma unroll
    for (int q = 0; q < 4; ++q) {
      const int e  = tid + q * 256;
      const int b  = e >> 4;
      const int jl = e & 15;
      float gi = gatesL[b * 68 +  0 + jl] + xv[q][0];
      float gf = gatesL[b * 68 + 16 + jl] + xv[q][1];
      float gg = gatesL[b * 68 + 32 + jl] + xv[q][2];
      float go = gatesL[b * 68 + 48 + jl] + xv[q][3];
      gi = sigmoidf_fast(gi);
      gf = sigmoidf_fast(gf);
      go = sigmoidf_fast(go);
      gg = tanhf_fast(gg);
      c[q] = gf * c[q] + gi * gg;
      const float hv = go * tanhf_fast(c[q]);
      hnext[(size_t)b * H_ + j0 + jl] = (bf16)hv;
      if (t == T_ - 1) out[(size_t)b * H_ + j0 + jl] = hv;
    }
    __syncthreads();   // gatesL reused next step
    __threadfence();   // publish hnext
    grid_barrier(bar, tid);
  }
}

// ---------------------------------------------------------------- launcher
extern "C" void kernel_launch(void* const* d_in, const int* in_sizes, int n_in,
                              void* d_out, int out_size, void* d_ws, size_t ws_size,
                              hipStream_t stream) {
  const float* x   = (const float*)d_in[0];   // [B,T,I]
  const float* Wih = (const float*)d_in[1];   // [4H,I]
  const float* Whh = (const float*)d_in[2];   // [4H,H]
  const float* bih = (const float*)d_in[3];   // [4H]
  const float* bhh = (const float*)d_in[4];   // [4H]
  float* out = (float*)d_out;                 // [B,H]

  char* ws = (char*)d_ws;
  bf16* xb   = (bf16*)ws;                                      // 64 MiB
  bf16* hbuf = (bf16*)(ws + (size_t)67108864);                 // 128 KiB (double buffer)
  int*  bar  = (int*) (ws + (size_t)67108864 + 131072);        // barrier state
  float* xg  = (float*)(ws + (size_t)67108864 + 131072 + 256); // 512 MiB

  lstm_cvt_x<<<2048, 256, 0, stream>>>(x, xb, B_ * T_ * I_);
  lstm_init_bar<<<1, 32, 0, stream>>>(bar);

  dim3 g1(65536 / 256, G_ / 64);   // 256 x 32 blocks
  lstm_xg_gemm<<<g1, 256, 65536, stream>>>(xb, Wih, bih, bhh, xg);

  lstm_scan<<<NWG, 256, 65536 + 64 * 68 * 4, stream>>>(Whh, xg, hbuf, out, bar);
}
